// OtterLMBlock_18219251270272
// MI455X (gfx1250) — compile-verified, matrix operand order checked by reading, not run on hardware
//
#include <hip/hip_runtime.h>

// ---------------------------------------------------------------------------
// Types for CDNA5 WMMA (wave32, v_wmma_f32_16x16x32_bf16)
// ---------------------------------------------------------------------------
typedef __bf16        bf16x16 __attribute__((ext_vector_type(16)));
typedef float         f32x8   __attribute__((ext_vector_type(8)));
typedef unsigned int  u32x4   __attribute__((ext_vector_type(4)));

union Frag16 {
    bf16x16 bf;
    u32x4   u[2];
};

__device__ __forceinline__ f32x8 wmma_bf16(const Frag16& a, const Frag16& b, f32x8 c) {
    return __builtin_amdgcn_wmma_f32_16x16x32_bf16(
        /*neg_a=*/false, a.bf, /*neg_b=*/false, b.bf,
        /*c_mod=*/(short)0, c, /*reuse_a=*/false, /*reuse_b=*/false);
}

// CDNA5 async global->LDS copy (ASYNCcnt-tracked, 16B per lane).
// LDS offset = low 32 bits of the generic pointer (LDS aperture maps addr[31:0]).
__device__ __forceinline__ void async_b128(const void* g, void* ldsPtr) {
    unsigned long long ga  = (unsigned long long)g;
    unsigned           off = (unsigned)(unsigned long long)ldsPtr;
    asm volatile("global_load_async_to_lds_b128 %0, %1, off"
                 :: "v"(off), "v"(ga) : "memory");
}
__device__ __forceinline__ void wait_async0() {
    asm volatile("s_wait_asynccnt 0x0" ::: "memory");
}

// Reductions within a 16-lane half-wave (C/D rows live across 16 lanes).
__device__ __forceinline__ float halfMax16(float v) {
#pragma unroll
    for (int m = 1; m < 16; m <<= 1) v = fmaxf(v, __shfl_xor(v, m, 32));
    return v;
}
__device__ __forceinline__ float halfSum16(float v) {
#pragma unroll
    for (int m = 1; m < 16; m <<= 1) v += __shfl_xor(v, m, 32);
    return v;
}

// ---------------------------------------------------------------------------
// Constants for this block
// ---------------------------------------------------------------------------
#define BB   4
#define TT   1024
#define CC   2048
#define HH   32
#define HKV  8
#define HD   64
#define HID  5632
#define NQKV 3072   // (H + 2*HKV) * HD

// ---------------------------------------------------------------------------
// Weight transpose + cast:  in  f32 (K x N)  ->  out bf16 (N x K)
// ---------------------------------------------------------------------------
__global__ __launch_bounds__(256)
void transpose_cast(const float* __restrict__ in, __bf16* __restrict__ out,
                    int K, int N) {
    __shared__ float tile[32][33];
    const int tx = threadIdx.x & 31;
    const int ty = threadIdx.x >> 5;          // 0..7
    const int n0 = blockIdx.x * 32;
    const int k0 = blockIdx.y * 32;
#pragma unroll
    for (int j = 0; j < 32; j += 8)
        tile[ty + j][tx] = in[(size_t)(k0 + ty + j) * N + n0 + tx];
    __syncthreads();
#pragma unroll
    for (int j = 0; j < 32; j += 8)
        out[(size_t)(n0 + ty + j) * K + k0 + tx] = (__bf16)tile[tx][ty + j];
}

// ---------------------------------------------------------------------------
// RMSNorm: per row of length C=2048, writes bf16 normalized row.
// ---------------------------------------------------------------------------
__global__ __launch_bounds__(256)
void rmsnorm_kernel(const float* __restrict__ x, const float* __restrict__ scale,
                    __bf16* __restrict__ out) {
    const int row = blockIdx.x;
    const float* xr = x + (size_t)row * CC;
    float ss = 0.f;
#pragma unroll
    for (int i = threadIdx.x; i < CC; i += 256) { float v = xr[i]; ss += v * v; }
#pragma unroll
    for (int m = 1; m < 32; m <<= 1) ss += __shfl_xor(ss, m, 32);
    __shared__ float red[8];
    if ((threadIdx.x & 31) == 0) red[threadIdx.x >> 5] = ss;
    __syncthreads();
    float tot = 0.f;
#pragma unroll
    for (int w = 0; w < 8; ++w) tot += red[w];
    const float inv = rsqrtf(tot * (1.0f / CC) + 1e-6f);
#pragma unroll
    for (int i = threadIdx.x; i < CC; i += 256)
        out[(size_t)row * CC + i] = (__bf16)(xr[i] * scale[i] * inv);
}

// ---------------------------------------------------------------------------
// bf16 WMMA GEMM:  C(MxN,f32) = A(MxK,bf16 rowmajor) * Bt(NxK,bf16 rowmajor)^T
// Block tile 128x128, BK=32, 8 waves (4x2 wave grid), per-wave 32x64.
// LDS double-buffered; tiles staged with global_load_async_to_lds_b128 so the
// next K-tile streams into LDS while WMMAs consume the current one.
// EPI==1: C += Res (f32 residual).
// ---------------------------------------------------------------------------
template <int EPI>
__global__ __launch_bounds__(256)
void gemm_bf16(const __bf16* __restrict__ A, const __bf16* __restrict__ Bt,
               const float* __restrict__ Res, float* __restrict__ Cout,
               int M, int N, int K) {
    constexpr int LDT = 40;                   // 32 + 8 pad (80B rows, 16B aligned)
    __shared__ __bf16 lA[2][128 * LDT];
    __shared__ __bf16 lB[2][128 * LDT];

    const int tid  = threadIdx.x;
    const int lane = tid & 31;
    const int wave = tid >> 5;
    const int wm   = wave >> 1;               // 0..3
    const int wn   = wave & 1;                // 0..1
    const int rowBlk = blockIdx.y * 128;
    const int colBlk = blockIdx.x * 128;
    const int m16 = lane & 15;
    const int hi  = lane >> 4;
    const int klo = hi << 3;                  // 0 or 8 (K base per ISA layout)

    auto stage = [&](int k0, int bufi) {
#pragma unroll
        for (int c = tid; c < 512; c += 256) {
            const int r  = c >> 2;
            const int cc = (c & 3) << 3;
            async_b128(&A[(size_t)(rowBlk + r) * K + k0 + cc],
                       &lA[bufi][r * LDT + cc]);
            async_b128(&Bt[(size_t)(colBlk + r) * K + k0 + cc],
                       &lB[bufi][r * LDT + cc]);
        }
    };

    f32x8 acc[2][4] = {};

    stage(0, 0);
    int buf = 0;
    for (int k0 = 0; k0 < K; k0 += 32) {
        wait_async0();                        // own async tile complete
        __syncthreads();                      // everyone's tile complete
        if (k0 + 32 < K) stage(k0 + 32, buf ^ 1);

        Frag16 a[2], b[4];
#pragma unroll
        for (int mi = 0; mi < 2; ++mi) {
            const __bf16* p = &lA[buf][(wm * 32 + mi * 16 + m16) * LDT + klo];
            a[mi].u[0] = *(const u32x4*)p;
            a[mi].u[1] = *(const u32x4*)(p + 16);
        }
#pragma unroll
        for (int ni = 0; ni < 4; ++ni) {
            const __bf16* p = &lB[buf][(wn * 64 + ni * 16 + m16) * LDT + klo];
            b[ni].u[0] = *(const u32x4*)p;
            b[ni].u[1] = *(const u32x4*)(p + 16);
        }
#pragma unroll
        for (int mi = 0; mi < 2; ++mi)
#pragma unroll
            for (int ni = 0; ni < 4; ++ni)
                acc[mi][ni] = wmma_bf16(a[mi], b[ni], acc[mi][ni]);
        buf ^= 1;
    }

    const int rbase = hi << 3;                // lanes>=16 hold rows M+8
#pragma unroll
    for (int mi = 0; mi < 2; ++mi)
#pragma unroll
        for (int ni = 0; ni < 4; ++ni)
#pragma unroll
            for (int r = 0; r < 8; ++r) {
                const int row = rowBlk + wm * 32 + mi * 16 + r + rbase;
                const int col = colBlk + wn * 64 + ni * 16 + m16;
                float v = acc[mi][ni][r];
                if (EPI == 1) v += Res[(size_t)row * N + col];
                Cout[(size_t)row * N + col] = v;
            }
}

// ---------------------------------------------------------------------------
// RoPE + repack: qkv f32 (B*T, 3072) ->
//   qpk bf16 [B][H][T][HD]  (pre-scaled by 1/sqrt(HD))
//   kpk bf16 [B][HKV][T][HD]
//   vt  bf16 [B][HKV][HD][T]   (transposed for the PV WMMA B-operand)
// ---------------------------------------------------------------------------
__global__ __launch_bounds__(256)
void rope_pack(const float* __restrict__ qkv,
               const float* __restrict__ cosT, const float* __restrict__ sinT,
               __bf16* __restrict__ qpk, __bf16* __restrict__ kpk,
               __bf16* __restrict__ vt) {
    const int bt = blockIdx.x;
    const int b = bt >> 10, t = bt & 1023;
    const float* row = qkv + (size_t)bt * NQKV;
#pragma unroll
    for (int p = threadIdx.x; p < 1536; p += 256) {
        const int head = p >> 5;              // 0..47
        const int d2   = p & 31;              // pair index
        const float e = row[head * HD + 2 * d2];
        const float o = row[head * HD + 2 * d2 + 1];
        if (head < HH) {
            const float c = cosT[t * 32 + d2], s = sinT[t * 32 + d2];
            const float re = e * c - o * s, ro = e * s + o * c;
            const size_t base = (((size_t)b * HH + head) * TT + t) * HD + 2 * d2;
            qpk[base]     = (__bf16)(re * 0.125f);
            qpk[base + 1] = (__bf16)(ro * 0.125f);
        } else if (head < HH + HKV) {
            const int kh = head - HH;
            const float c = cosT[t * 32 + d2], s = sinT[t * 32 + d2];
            const float re = e * c - o * s, ro = e * s + o * c;
            const size_t base = (((size_t)b * HKV + kh) * TT + t) * HD + 2 * d2;
            kpk[base]     = (__bf16)re;
            kpk[base + 1] = (__bf16)ro;
        } else {
            const int vh = head - HH - HKV;
            const size_t base = (((size_t)b * HKV + vh) * HD + 2 * d2) * TT + t;
            vt[base]      = (__bf16)e;
            vt[base + TT] = (__bf16)o;
        }
    }
}

// ---------------------------------------------------------------------------
// Flash attention (causal, GQA 4:1). Block = 4 waves, each wave owns 16 q rows
// of a 64-row q tile. KV tiles of 64 double-buffered in LDS via async copies;
// online softmax; S = Q*K^T and O += P*V via v_wmma_f32_16x16x32_bf16.
// ---------------------------------------------------------------------------
__global__ __launch_bounds__(128)
void attn_kernel(const __bf16* __restrict__ qpk, const __bf16* __restrict__ kpk,
                 const __bf16* __restrict__ vt, __bf16* __restrict__ ybf) {
    constexpr int LDK = 72;                   // 64 + 8 pad (144B rows)
    __shared__ __bf16 lK[2][64 * LDK];        // K tile: rows t, contiguous d
    __shared__ __bf16 lV[2][64 * LDK];        // V^T tile: rows d, contiguous t
    __shared__ __bf16 lP[4][16 * LDK];        // per-wave P scratch (16 x 64)

    const int qtile = blockIdx.x;
    const int head  = blockIdx.y;
    const int b     = blockIdx.z;
    const int kvh   = head >> 2;              // n_rep = 4

    const __bf16* Q  = qpk + (((size_t)b * HH + head) * TT) * HD;
    const __bf16* Kp = kpk + (((size_t)b * HKV + kvh) * TT) * HD;
    const __bf16* Vt = vt  + (((size_t)b * HKV + kvh) * HD) * TT;

    const int tid  = threadIdx.x;
    const int lane = tid & 31;
    const int wave = tid >> 5;
    const int m16  = lane & 15;
    const int hi   = lane >> 4;
    const int klo  = hi << 3;
    const int q0   = qtile * 64 + wave * 16;  // this wave's first q row

    auto stageKV = [&](int j, int bufi) {
        const int jb = j * 64;
#pragma unroll
        for (int c = tid; c < 512; c += 128) {
            const int r  = c >> 3;
            const int cc = (c & 7) << 3;
            async_b128(&Kp[(size_t)(jb + r) * HD + cc], &lK[bufi][r * LDK + cc]);
            async_b128(&Vt[(size_t)r * TT + jb + cc],   &lV[bufi][r * LDK + cc]);
        }
    };

    // Q fragments (16 x 64, two K-steps), loaded once from global.
    Frag16 aq[2];
#pragma unroll
    for (int ks = 0; ks < 2; ++ks) {
        const __bf16* p = &Q[(size_t)(q0 + m16) * HD + ks * 32 + klo];
        aq[ks].u[0] = *(const u32x4*)p;
        aq[ks].u[1] = *(const u32x4*)(p + 16);
    }

    f32x8 o[4] = {};
    float mx[8], ls[8];
#pragma unroll
    for (int r = 0; r < 8; ++r) { mx[r] = -1e30f; ls[r] = 0.f; }

    stageKV(0, 0);
    int buf = 0;
    for (int j = 0; j <= qtile; ++j) {
        const int jb = j * 64;
        wait_async0();
        __syncthreads();
        if (j < qtile) stageKV(j + 1, buf ^ 1);

        // S = Q * K^T  (16 x 64, scale folded into Q)
        f32x8 s[4] = {};
#pragma unroll
        for (int ks = 0; ks < 2; ++ks) {
#pragma unroll
            for (int ni = 0; ni < 4; ++ni) {
                Frag16 bk;
                const __bf16* p = &lK[buf][(ni * 16 + m16) * LDK + ks * 32 + klo];
                bk.u[0] = *(const u32x4*)p;
                bk.u[1] = *(const u32x4*)(p + 16);
                s[ni] = wmma_bf16(aq[ks], bk, s[ni]);
            }
        }

        // causal mask on diagonal tile only
        if (j == qtile) {
#pragma unroll
            for (int ni = 0; ni < 4; ++ni)
#pragma unroll
                for (int r = 0; r < 8; ++r) {
                    const int trow = q0 + r + (hi << 3);
                    const int tcol = jb + ni * 16 + m16;
                    if (tcol > trow) s[ni][r] = -1e30f;
                }
        }

        // online softmax update per row
#pragma unroll
        for (int r = 0; r < 8; ++r) {
            float tm = s[0][r];
#pragma unroll
            for (int ni = 1; ni < 4; ++ni) tm = fmaxf(tm, s[ni][r]);
            tm = halfMax16(tm);
            const float mnew  = fmaxf(mx[r], tm);
            const float alpha = __expf(mx[r] - mnew);
            float ts = 0.f;
#pragma unroll
            for (int ni = 0; ni < 4; ++ni) {
                const float pv = __expf(s[ni][r] - mnew);
                s[ni][r] = pv;
                ts += pv;
            }
            ts = halfSum16(ts);
            ls[r] = ls[r] * alpha + ts;
            mx[r] = mnew;
#pragma unroll
            for (int ni = 0; ni < 4; ++ni) o[ni][r] *= alpha;
        }

        // P (C-layout) -> bf16 A-layout via per-wave LDS scratch
        __bf16* Pw = &lP[wave][0];
#pragma unroll
        for (int ni = 0; ni < 4; ++ni)
#pragma unroll
            for (int r = 0; r < 8; ++r)
                Pw[(r + (hi << 3)) * LDK + ni * 16 + m16] = (__bf16)s[ni][r];
        // same-wave LDS ops are in-order (DScnt) -> safe to read back

        // O += P * V
#pragma unroll
        for (int ks = 0; ks < 2; ++ks) {
            Frag16 ap;
            const __bf16* pp = &Pw[m16 * LDK + ks * 32 + klo];
            ap.u[0] = *(const u32x4*)pp;
            ap.u[1] = *(const u32x4*)(pp + 16);
#pragma unroll
            for (int ni = 0; ni < 4; ++ni) {
                Frag16 bv;
                const __bf16* pv = &lV[buf][(ni * 16 + m16) * LDK + ks * 32 + klo];
                bv.u[0] = *(const u32x4*)pv;
                bv.u[1] = *(const u32x4*)(pv + 16);
                o[ni] = wmma_bf16(ap, bv, o[ni]);
            }
        }
        buf ^= 1;
    }

    // normalize and write y as bf16 (B,T,C) row-major
#pragma unroll
    for (int ni = 0; ni < 4; ++ni)
#pragma unroll
        for (int r = 0; r < 8; ++r) {
            const int t   = q0 + r + (hi << 3);
            const int col = head * HD + ni * 16 + m16;
            ybf[((size_t)b * TT + t) * CC + col] = (__bf16)(o[ni][r] / ls[r]);
        }
}

// ---------------------------------------------------------------------------
// act = silu(g) * u, cast to bf16 (A-operand of down projection)
// ---------------------------------------------------------------------------
__global__ __launch_bounds__(256)
void silu_mul(const float* __restrict__ g, const float* __restrict__ u,
              __bf16* __restrict__ out, size_t n) {
    for (size_t i = (size_t)blockIdx.x * blockDim.x + threadIdx.x; i < n;
         i += (size_t)gridDim.x * blockDim.x) {
        const float gv = g[i];
        const float sg = gv / (1.f + __expf(-gv));
        out[i] = (__bf16)(sg * u[i]);
    }
}

// ---------------------------------------------------------------------------
// Host launcher
// ---------------------------------------------------------------------------
extern "C" void kernel_launch(void* const* d_in, const int* in_sizes, int n_in,
                              void* d_out, int out_size, void* d_ws, size_t ws_size,
                              hipStream_t stream) {
    const float* x      = (const float*)d_in[0];
    const float* cosT   = (const float*)d_in[1];
    const float* sinT   = (const float*)d_in[2];
    const float* w_attn = (const float*)d_in[3];
    const float* w_proj = (const float*)d_in[4];
    const float* w_gate = (const float*)d_in[5];
    const float* w_up   = (const float*)d_in[6];
    const float* w_down = (const float*)d_in[7];
    const float* scale1 = (const float*)d_in[8];
    const float* scale2 = (const float*)d_in[9];
    float* out = (float*)d_out;

    char* p = (char*)d_ws;
    auto carve = [&](size_t bytes) -> void* {
        void* r = (void*)p;
        p += (bytes + 255) & ~(size_t)255;
        return r;
    };

    const size_t NT = (size_t)BB * TT;                     // 4096 token rows
    __bf16* wtA  = (__bf16*)carve((size_t)NQKV * CC * 2);  // (3072 x 2048)
    __bf16* wtP  = (__bf16*)carve((size_t)CC * CC * 2);
    __bf16* wtG  = (__bf16*)carve((size_t)HID * CC * 2);
    __bf16* wtU  = (__bf16*)carve((size_t)HID * CC * 2);
    __bf16* wtD  = (__bf16*)carve((size_t)CC * HID * 2);
    __bf16* h1   = (__bf16*)carve(NT * CC * 2);
    float*  qkv  = (float*) carve(NT * NQKV * 4);
    __bf16* qpk  = (__bf16*)carve((size_t)BB * HH * TT * HD * 2);
    __bf16* kpk  = (__bf16*)carve((size_t)BB * HKV * TT * HD * 2);
    __bf16* vtb  = (__bf16*)carve((size_t)BB * HKV * TT * HD * 2);
    __bf16* ybf  = (__bf16*)carve(NT * CC * 2);
    float*  x2   = (float*) carve(NT * CC * 4);
    __bf16* h2   = (__bf16*)carve(NT * CC * 2);
    float*  gbuf = (float*) carve(NT * HID * 4);
    float*  ubuf = (float*) carve(NT * HID * 4);
    __bf16* actb = (__bf16*)carve(NT * HID * 2);

    // 1. weights -> bf16 W^T (N x K)
    transpose_cast<<<dim3(NQKV / 32, CC / 32), 256, 0, stream>>>(w_attn, wtA, CC, NQKV);
    transpose_cast<<<dim3(CC / 32, CC / 32),   256, 0, stream>>>(w_proj, wtP, CC, CC);
    transpose_cast<<<dim3(HID / 32, CC / 32),  256, 0, stream>>>(w_gate, wtG, CC, HID);
    transpose_cast<<<dim3(HID / 32, CC / 32),  256, 0, stream>>>(w_up,   wtU, CC, HID);
    transpose_cast<<<dim3(CC / 32, HID / 32),  256, 0, stream>>>(w_down, wtD, HID, CC);

    // 2. h1 = rmsnorm(x, scale1)
    rmsnorm_kernel<<<NT, 256, 0, stream>>>(x, scale1, h1);

    // 3. qkv = h1 @ w_attn
    gemm_bf16<0><<<dim3(NQKV / 128, NT / 128), 256, 0, stream>>>(
        h1, wtA, nullptr, qkv, NT, NQKV, CC);

    // 4. rotary + pack (Q pre-scaled, V transposed)
    rope_pack<<<NT, 256, 0, stream>>>(qkv, cosT, sinT, qpk, kpk, vtb);

    // 5. causal GQA flash attention -> y (bf16)
    attn_kernel<<<dim3(TT / 64, HH, BB), 128, 0, stream>>>(qpk, kpk, vtb, ybf);

    // 6. x2 = x + y @ w_proj
    gemm_bf16<1><<<dim3(CC / 128, NT / 128), 256, 0, stream>>>(
        ybf, wtP, x, x2, NT, CC, CC);

    // 7. h2 = rmsnorm(x2, scale2)
    rmsnorm_kernel<<<NT, 256, 0, stream>>>(x2, scale2, h2);

    // 8. gate / up projections
    gemm_bf16<0><<<dim3(HID / 128, NT / 128), 256, 0, stream>>>(
        h2, wtG, nullptr, gbuf, NT, HID, CC);
    gemm_bf16<0><<<dim3(HID / 128, NT / 128), 256, 0, stream>>>(
        h2, wtU, nullptr, ubuf, NT, HID, CC);

    // 9. act = silu(g) * u  (bf16)
    silu_mul<<<2048, 256, 0, stream>>>(gbuf, ubuf, actb, NT * (size_t)HID);

    // 10. out = x2 + act @ w_down
    gemm_bf16<1><<<dim3(CC / 128, NT / 128), 256, 0, stream>>>(
        actb, wtD, x2, out, NT, CC, HID);
}
